// FactorizedTransition_6262062318073
// MI455X (gfx1250) — matrix operand order repaired and checked by reading, not compiled, not run on hardware
//
#include <hip/hip_runtime.h>
#include <hip/hip_bf16.h>

// ---------------------------------------------------------------------------
// Fused FactorizedTransition for MI455X (gfx1250, wave32, WMMA bf16)
//   prior = belief @ softmax(q k^T),  q = emb@Wq+bq, k = emb@Wk+bk
// Flash-style: never materialize the 8192x8192 T matrix (256MB).
//   pass 1: per-row max + sum-exp stats (recompute S tiles via WMMA)
//   pass 2: recompute S, form P=exp(s-m)/l in bf16, acc += belief @ P (WMMA)
// All matmuls use v_wmma_f32_16x16x32_bf16 with fp32 accumulation.
// kb tiles are staged into LDS with GLOBAL_LOAD_ASYNC_TO_LDS_B128 (ASYNCcnt)
// so waves share one global fetch per workgroup (double-buffered in pass 1,
// loop-invariant single load in pass 2).
// ---------------------------------------------------------------------------

typedef __attribute__((ext_vector_type(16))) __bf16 bf16x16;
typedef __attribute__((ext_vector_type(8)))  __bf16 bf16x8;
typedef __attribute__((ext_vector_type(8)))  float  f32x8;
typedef __attribute__((ext_vector_type(4)))  float  f32x4;
typedef int v4i_ __attribute__((vector_size(16)));   // builtin's pointee type

#define NN 8192
#define EE 128
#define HH 128
#define BB 64
#define KPAD 136   // LDS row stride for 128-wide kb rows: 16B aligned, bank-spread

#if defined(__has_builtin)
#if __has_builtin(__builtin_amdgcn_global_load_async_to_lds_b128)
#define HAVE_ASYNC_LDS 1
#endif
#endif

__device__ __forceinline__ __bf16 f2bf(float f) {
  unsigned u = __builtin_bit_cast(unsigned, f);
  unsigned r = u + 0x7FFFu + ((u >> 16) & 1u);   // round-to-nearest-even
  unsigned short h = (unsigned short)(r >> 16);
  return __builtin_bit_cast(__bf16, h);
}

// A fragment (16x32 bf16, row-major source, leading dim ld):
// lane L (m=L&15, hi=L>>4) holds row m:  K = k0+hi*8+{0..7} in v[0..3],
// K = k0+16+hi*8+{0..7} in v[4..7]   (ISA 16-bit A layout)
__device__ __forceinline__ bf16x16 load_a_frag(const __bf16* base, int row0,
                                               int k0, int ld) {
  int lane = threadIdx.x & 31;
  int m = lane & 15, hi = lane >> 4;
  const __bf16* p = base + (size_t)(row0 + m) * ld + k0 + hi * 8;
  bf16x8 lo = *(const bf16x8*)p;
  bf16x8 hh = *(const bf16x8*)(p + 16);
  bf16x16 r;
#pragma unroll
  for (int t = 0; t < 8; ++t) { r[t] = lo[t]; r[t + 8] = hh[t]; }
  return r;
}

// B fragment (32x16 bf16). Source stored "transposed": row j holds all K for
// column j, i.e. B[K][n] = base[(col0+n)*ld + k0 + K].
// lane L (n=L&15, hi=L>>4) holds column n: K = k0+hi*8+{0..7} in v[0..3],
// K = k0+16+hi*8+{0..7} in v[4..7]
// Works for both global and (inlined) LDS pointers.
__device__ __forceinline__ bf16x16 load_b_frag(const __bf16* base, int col0,
                                               int k0, int ld) {
  int lane = threadIdx.x & 31;
  int n = lane & 15, hi = lane >> 4;
  const __bf16* p = base + (size_t)(col0 + n) * ld + k0 + hi * 8;
  bf16x8 lo = *(const bf16x8*)p;
  bf16x8 hh = *(const bf16x8*)(p + 16);
  bf16x16 r;
#pragma unroll
  for (int t = 0; t < 8; ++t) { r[t] = lo[t]; r[t + 8] = hh[t]; }
  return r;
}

__device__ __forceinline__ f32x8 wmma_bf16(bf16x16 a, bf16x16 b, f32x8 c) {
  return __builtin_amdgcn_wmma_f32_16x16x32_bf16(false, a, false, b,
                                                 (short)0, c, false, false);
}

// Stage 64 rows x 128 cols of kb (rows j0..j0+63) into LDS with row stride
// KPAD.  1024 16B chunks, 128 threads x 8 chunks.  Async (ASYNCcnt) when the
// builtin exists; plain LDS vector stores otherwise.
__device__ __forceinline__ void stage_k_strip(const __bf16* __restrict__ kb,
                                              __bf16* lds, int j0) {
  int t = threadIdx.x;  // 0..127
#pragma unroll
  for (int c = 0; c < 8; ++c) {
    int id = t + c * 128;            // 0..1023
    int row = id >> 4, cb = id & 15; // 64 rows x 16 chunks of 8 bf16
    const __bf16* src = kb + (size_t)(j0 + row) * HH + cb * 8;
    __bf16* dst = lds + row * KPAD + cb * 8;
#if HAVE_ASYNC_LDS
    __builtin_amdgcn_global_load_async_to_lds_b128(
        (__attribute__((address_space(1))) v4i_*)src,
        (__attribute__((address_space(3))) v4i_*)dst, 0, 0);
#else
    *(bf16x8*)dst = *(const bf16x8*)src;
#endif
  }
}

__device__ __forceinline__ void wait_async_done() {
#if HAVE_ASYNC_LDS
#if __has_builtin(__builtin_amdgcn_s_wait_asynccnt)
  __builtin_amdgcn_s_wait_asynccnt(0);
#else
  asm volatile("s_wait_asynccnt 0" ::: "memory");
#endif
#endif
}

// ---------------------------------------------------------------------------
// Kernel 0: fp32 -> bf16 conversions (+ transpose W so B-frag loads are
// two contiguous 16B loads per lane)
// ---------------------------------------------------------------------------
__global__ void k_convert(const float* __restrict__ belief,
                          const float* __restrict__ emb,
                          const float* __restrict__ Wq,
                          const float* __restrict__ Wk,
                          __bf16* __restrict__ belief_bf,
                          __bf16* __restrict__ emb_bf,
                          __bf16* __restrict__ WqT,
                          __bf16* __restrict__ WkT) {
  int stride = gridDim.x * blockDim.x;
  int idx = blockIdx.x * blockDim.x + threadIdx.x;
  for (int i = idx; i < NN * EE; i += stride) emb_bf[i] = f2bf(emb[i]);
  for (int i = idx; i < BB * NN; i += stride) belief_bf[i] = f2bf(belief[i]);
  for (int i = idx; i < EE * HH; i += stride) {
    int e = i / HH, h = i % HH;                 // Wq is [E,H] row-major
    WqT[h * EE + e] = f2bf(Wq[i]);              // WqT is [H,E]
    WkT[h * EE + e] = f2bf(Wk[i]);
  }
}

// ---------------------------------------------------------------------------
// Kernel 1: q = emb@Wq + bq ; k = emb@Wk + bk   (bf16 outputs, [N,128])
// one wave per 16-row tile; 4 K-frags x 8 H-tiles x 2 matrices
// ---------------------------------------------------------------------------
__global__ __launch_bounds__(128) void k_qk(const __bf16* __restrict__ emb_bf,
                                            const __bf16* __restrict__ WqT,
                                            const __bf16* __restrict__ WkT,
                                            const float* __restrict__ bq,
                                            const float* __restrict__ bk,
                                            __bf16* __restrict__ qb,
                                            __bf16* __restrict__ kb) {
  int wave = (blockIdx.x * blockDim.x + threadIdx.x) >> 5;
  int i0 = wave * 16;
  if (i0 >= NN) return;
  int lane = threadIdx.x & 31, n = lane & 15, hi = lane >> 4;
  bf16x16 a[4];
#pragma unroll
  for (int kk = 0; kk < 4; ++kk) a[kk] = load_a_frag(emb_bf, i0, kk * 32, EE);
#pragma unroll
  for (int mat = 0; mat < 2; ++mat) {
    const __bf16* W = mat ? WkT : WqT;
    const float* bias = mat ? bk : bq;
    __bf16* out = mat ? kb : qb;
#pragma unroll
    for (int ht = 0; ht < 8; ++ht) {
      f32x8 c = {};
#pragma unroll
      for (int kk = 0; kk < 4; ++kk)
        c = wmma_bf16(a[kk], load_b_frag(W, ht * 16, kk * 32, EE), c);
      float bv = bias[ht * 16 + n];
      __bf16* op = out + (size_t)(i0 + hi * 8) * HH + ht * 16 + n;
#pragma unroll
      for (int r = 0; r < 8; ++r) op[r * HH] = f2bf(c[r] + bv);
    }
  }
}

// ---------------------------------------------------------------------------
// Kernel 2: per-row softmax stats. 4 waves/workgroup, each owns 16 rows of S;
// the workgroup sweeps j in 64-wide strips staged into double-buffered LDS
// (async copy of strip n+1 overlapped with WMMA+softmax math on strip n).
// Online max/sum-exp per lane slot, cross-lane merge with shfl_xor (wave32),
// writes m[i] and 1/l[i].
// ---------------------------------------------------------------------------
__global__ __launch_bounds__(128) void k_rowstats(const __bf16* __restrict__ qb,
                                                  const __bf16* __restrict__ kb,
                                                  float* __restrict__ mrow,
                                                  float* __restrict__ rrow) {
  __shared__ __attribute__((aligned(16))) __bf16 KS[2][64 * KPAD];
  int wave = (blockIdx.x * blockDim.x + threadIdx.x) >> 5;   // grid-exact
  int i0 = wave * 16;
  int lane = threadIdx.x & 31, n = lane & 15, hi = lane >> 4;
  bf16x16 a[4];
#pragma unroll
  for (int kk = 0; kk < 4; ++kk) a[kk] = load_a_frag(qb, i0, kk * 32, HH);
  float mr[8], lr[8];
#pragma unroll
  for (int r = 0; r < 8; ++r) { mr[r] = -3.0e38f; lr[r] = 0.0f; }

  stage_k_strip(kb, &KS[0][0], 0);       // preload first strip
  wait_async_done();
  __syncthreads();

  for (int j0 = 0; j0 < NN; j0 += 64) {
    int cur = (j0 >> 6) & 1;
    if (j0 + 64 < NN)
      stage_k_strip(kb, &KS[cur ^ 1][0], j0 + 64);   // prefetch next strip
    f32x8 s[4];
#pragma unroll
    for (int jt = 0; jt < 4; ++jt) {
      f32x8 c = {};
#pragma unroll
      for (int kk = 0; kk < 4; ++kk)
        c = wmma_bf16(a[kk], load_b_frag(&KS[cur][0], jt * 16, kk * 32, KPAD), c);
      s[jt] = c;
    }
#pragma unroll
    for (int r = 0; r < 8; ++r) {
      float tm = fmaxf(fmaxf(s[0][r], s[1][r]), fmaxf(s[2][r], s[3][r]));
      float nm = fmaxf(mr[r], tm);
      float acc = lr[r] * __expf(mr[r] - nm);
      acc += __expf(s[0][r] - nm) + __expf(s[1][r] - nm) +
             __expf(s[2][r] - nm) + __expf(s[3][r] - nm);
      lr[r] = acc;
      mr[r] = nm;
    }
    wait_async_done();
    __syncthreads();
  }
  // merge partial (m,l) across the 16 lanes sharing each row (xor<16 keeps hi)
#pragma unroll
  for (int mask = 1; mask <= 8; mask <<= 1) {
#pragma unroll
    for (int r = 0; r < 8; ++r) {
      float om = __shfl_xor(mr[r], mask, 32);
      float ol = __shfl_xor(lr[r], mask, 32);
      float nm = fmaxf(mr[r], om);
      lr[r] = lr[r] * __expf(mr[r] - nm) + ol * __expf(om - nm);
      mr[r] = nm;
    }
  }
  if (n == 0) {
#pragma unroll
    for (int r = 0; r < 8; ++r) {
      mrow[i0 + hi * 8 + r] = mr[r];
      rrow[i0 + hi * 8 + r] = 1.0f / lr[r];
    }
  }
}

// ---------------------------------------------------------------------------
// Kernel 3: prior[:, J0:J0+64] = sum_i belief[:, i] * P[i, j]
// Workgroup (4 waves) owns a 64-column strip.  kb rows J0..J0+63 are loop-
// invariant: staged ONCE into LDS (async), reused for all 256 i-iterations.
// Per 32-i step:
//   phase A: S (32x64) via WMMA from LDS kb tile, P = exp(s-m)*(1/l) -> bf16,
//            stored transposed into LDS (stride-40 rows, aligned b128)
//   phase B: acc[mb,jt] += belief_frag(16x32) x P_frag(32x16)   (WMMA K=32)
// ---------------------------------------------------------------------------
__global__ __launch_bounds__(128) void k_prior(const __bf16* __restrict__ qb,
                                               const __bf16* __restrict__ kb,
                                               const __bf16* __restrict__ belief_bf,
                                               const float* __restrict__ mrow,
                                               const float* __restrict__ rrow,
                                               float* __restrict__ out) {
  __shared__ __attribute__((aligned(16))) __bf16 KT[64 * KPAD];  // kb strip
  __shared__ __attribute__((aligned(16))) __bf16 PT[64 * 40];    // P transposed
  int w = threadIdx.x >> 5;                 // wave 0..3
  int lane = threadIdx.x & 31, n = lane & 15, hi = lane >> 4;
  int J0 = blockIdx.x * 64;
  int it = w >> 1;                          // this wave's 16-row i-subtile
  int jt0 = (w & 1) * 2;                    // this wave's pair of j-tiles
  f32x8 acc[4] = {{}, {}, {}, {}};

  stage_k_strip(kb, KT, J0);                // loop-invariant kb tile
  wait_async_done();
  __syncthreads();

  for (int i0 = 0; i0 < NN; i0 += 32) {
    // ---- phase A: S tile + softmax-normalize -> PT (transposed, bf16) ----
    bf16x16 aq[4];
#pragma unroll
    for (int kk = 0; kk < 4; ++kk)
      aq[kk] = load_a_frag(qb, i0 + it * 16, kk * 32, HH);
    const float* mp = mrow + i0 + it * 16 + hi * 8;   // this lane's 8 rows
    const float* rp = rrow + i0 + it * 16 + hi * 8;
    f32x4 m0 = *(const f32x4*)mp, m1 = *(const f32x4*)(mp + 4);
    f32x4 r0 = *(const f32x4*)rp, r1 = *(const f32x4*)(rp + 4);
    float mv[8] = {m0[0], m0[1], m0[2], m0[3], m1[0], m1[1], m1[2], m1[3]};
    float rv[8] = {r0[0], r0[1], r0[2], r0[3], r1[0], r1[1], r1[2], r1[3]};
#pragma unroll
    for (int t = 0; t < 2; ++t) {
      int jt = jt0 + t;
      f32x8 c = {};
#pragma unroll
      for (int kk = 0; kk < 4; ++kk)
        c = wmma_bf16(aq[kk], load_b_frag(KT, jt * 16, kk * 32, KPAD), c);
      bf16x8 pk;
#pragma unroll
      for (int r = 0; r < 8; ++r)
        pk[r] = f2bf(__expf(c[r] - mv[r]) * rv[r]);
      // C-frag lane holds 8 consecutive i for one j -> one aligned 16B store
      *(bf16x8*)&PT[(jt * 16 + n) * 40 + it * 16 + hi * 8] = pk;
    }
    __syncthreads();
    // ---- phase B: accumulate belief x P ----
    bf16x16 ab = load_a_frag(belief_bf, w * 16, i0, NN);
#pragma unroll
    for (int jt = 0; jt < 4; ++jt) {
      bf16x16 bp = load_b_frag(PT, jt * 16, 0, 40);
      acc[jt] = wmma_bf16(ab, bp, acc[jt]);
    }
    __syncthreads();
  }
  // ---- store fp32 output ----
#pragma unroll
  for (int jt = 0; jt < 4; ++jt) {
    float* op = out + (size_t)(w * 16 + hi * 8) * NN + J0 + jt * 16 + n;
#pragma unroll
    for (int r = 0; r < 8; ++r) op[r * NN] = acc[jt][r];
  }
}

// ---------------------------------------------------------------------------
// Host launcher
// ---------------------------------------------------------------------------
extern "C" void kernel_launch(void* const* d_in, const int* in_sizes, int n_in,
                              void* d_out, int out_size, void* d_ws, size_t ws_size,
                              hipStream_t stream) {
  const float* belief = (const float*)d_in[0];   // [B, N]
  const float* emb    = (const float*)d_in[1];   // [N, E]
  const float* Wq     = (const float*)d_in[2];   // [E, H]
  const float* bq     = (const float*)d_in[3];   // [H]
  const float* Wk     = (const float*)d_in[4];   // [E, H]
  const float* bk     = (const float*)d_in[5];   // [H]
  float* prior = (float*)d_out;                  // [B, N] fp32

  // workspace layout (all offsets 32B-aligned)
  char* ws = (char*)d_ws;
  __bf16* emb_bf    = (__bf16*)(ws + 0);                    // 2 MB
  __bf16* qb        = (__bf16*)(ws + 2097152);              // 2 MB
  __bf16* kb        = (__bf16*)(ws + 4194304);              // 2 MB
  __bf16* belief_bf = (__bf16*)(ws + 6291456);              // 1 MB
  __bf16* WqT       = (__bf16*)(ws + 7340032);              // 32 KB
  __bf16* WkT       = (__bf16*)(ws + 7372800);              // 32 KB
  float*  mrow      = (float*)(ws + 7405568);               // 32 KB
  float*  rrow      = (float*)(ws + 7438336);               // 32 KB
  if (ws_size < 7471104) return;                            // need ~7.2 MB

  k_convert<<<256, 256, 0, stream>>>(belief, emb, Wq, Wk,
                                     belief_bf, emb_bf, WqT, WkT);
  // 512 waves total for N/16 row tiles -> 128 blocks x 128 threads (4 waves)
  k_qk<<<128, 128, 0, stream>>>(emb_bf, WqT, WkT, bq, bk, qb, kb);
  k_rowstats<<<128, 128, 0, stream>>>(qb, kb, mrow, rrow);
  // 128 column strips of 64 -> one workgroup each
  k_prior<<<128, 128, 0, stream>>>(qb, kb, belief_bf, mrow, rrow, prior);
}
